// MyLinear_77249281786293
// MI455X (gfx1250) — compile-verified, matrix operand order checked by reading, not run on hardware
//
#include <hip/hip_runtime.h>
#include <stdint.h>

typedef __attribute__((ext_vector_type(16))) __bf16 v16bf;
typedef __attribute__((ext_vector_type(8)))  __bf16 v8bf;
typedef __attribute__((ext_vector_type(8)))  float  v8f;

#define M_DIM 8192
#define N_DIM 4096
#define K_DIM 4096
#define BM 256
#define BN 128
#define BK 32
#define LDE 40      // LDS row stride in bf16 elems (80B: 16B-aligned, bank-conflict-free reads)
#define KTILES (K_DIM / BK)

union Frag { v16bf v; v8bf h[2]; };

// Pack the high 16 bits (truncated-bf16) of two fp32 words into one dword:
// result = { hi16(b1) , hi16(b0) }  (b0 in low half) -- single v_perm_b32.
__device__ __forceinline__ uint32_t pack_hi16(uint32_t b0, uint32_t b1) {
    return __builtin_amdgcn_perm(b1, b0, 0x07060302u);
}
// Exact residual after truncating f to bf16 (compiler folds to v_fma_mix_f32_bf16).
__device__ __forceinline__ float lo_residual(float f, uint32_t bits) {
    return f - __builtin_bit_cast(float, bits & 0xFFFF0000u);
}

__global__ __launch_bounds__(256, 1)
void gemm_bf16x3_kernel(const float* __restrict__ X,
                        const float* __restrict__ W,
                        const float* __restrict__ Bias,
                        float* __restrict__ C) {
    // Double-buffered: 2*(2*256*40 + 2*128*40)*2B = 120 KB of the WGP's 320 KB.
    __shared__ __bf16 Ah[2][BM * LDE];
    __shared__ __bf16 Al[2][BM * LDE];
    __shared__ __bf16 Bh[2][BN * LDE];
    __shared__ __bf16 Bl[2][BN * LDE];

    const int tid  = threadIdx.x;
    const int lane = tid & 31;
    const int wave = tid >> 5;
    const int wm   = wave & 3;   // 4 waves along M, 64 rows each
    const int wn   = wave >> 2;  // 2 waves along N, 64 cols each
    const int m0   = blockIdx.y * BM;
    const int n0   = blockIdx.x * BN;

    // ---- global-load mapping (coalesced float4) ----
    const int arow = tid >> 3;          // A: 32 rows / pass, 8 passes (256 rows)
    const int acol = (tid & 7) * 4;     //    8 threads x float4 per 32-wide K row
    const int brow = tid >> 5;          // B: 8 k-rows / pass, 4 passes
    const int bcol = (tid & 31) * 4;    //    32 threads x float4 per 128-wide N row

    float4 ar[8], br[4];
    auto load_tiles = [&](int kc) {
        const float* xa = X + (size_t)m0 * K_DIM + (size_t)kc * BK;
        #pragma unroll
        for (int i = 0; i < 8; ++i)
            ar[i] = *(const float4*)(xa + (size_t)(arow + 32 * i) * K_DIM + acol);
        const float* wb = W + (size_t)(kc * BK) * N_DIM + n0;
        #pragma unroll
        for (int i = 0; i < 4; ++i)
            br[i] = *(const float4*)(wb + (size_t)(brow + 8 * i) * N_DIM + bcol);
    };

    auto stage_to_lds = [&](int bsel) {
        // A tile: [m][k], k-contiguous; hi/lo packed 2-per-dword with v_perm, b64 stores.
        #pragma unroll
        for (int i = 0; i < 8; ++i) {
            const uint32_t b0 = __builtin_bit_cast(uint32_t, ar[i].x);
            const uint32_t b1 = __builtin_bit_cast(uint32_t, ar[i].y);
            const uint32_t b2 = __builtin_bit_cast(uint32_t, ar[i].z);
            const uint32_t b3 = __builtin_bit_cast(uint32_t, ar[i].w);
            uint2 hp, lp;
            hp.x = pack_hi16(b0, b1);
            hp.y = pack_hi16(b2, b3);
            lp.x = pack_hi16(__builtin_bit_cast(uint32_t, lo_residual(ar[i].x, b0)),
                             __builtin_bit_cast(uint32_t, lo_residual(ar[i].y, b1)));
            lp.y = pack_hi16(__builtin_bit_cast(uint32_t, lo_residual(ar[i].z, b2)),
                             __builtin_bit_cast(uint32_t, lo_residual(ar[i].w, b3)));
            const int off = (arow + 32 * i) * LDE + acol;   // 8B-aligned
            *(uint2*)&Ah[bsel][off] = hp;
            *(uint2*)&Al[bsel][off] = lp;
        }
        // B tile: transpose to [n][k]; hi stored straight from the fp32 word's top
        // half (ds_store_b16_d16_hi -> zero VALU for hi), lo via single fma_mix.
        #pragma unroll
        for (int i = 0; i < 4; ++i) {
            const float f[4] = {br[i].x, br[i].y, br[i].z, br[i].w};
            const int kr = brow + 8 * i;
            #pragma unroll
            for (int j = 0; j < 4; ++j) {
                const uint32_t u = __builtin_bit_cast(uint32_t, f[j]);
                *(uint16_t*)&Bh[bsel][(bcol + j) * LDE + kr] = (uint16_t)(u >> 16);
                const float lf = lo_residual(f[j], u);
                *(uint16_t*)&Bl[bsel][(bcol + j) * LDE + kr] =
                    (uint16_t)(__builtin_bit_cast(uint32_t, lf) >> 16);
            }
        }
    };

    // ---- fragment addressing per ISA 16-bit A/B layout ----
    // lane<16: row/col = +lane,    K = 0..7 then 16..23
    // lane>=16: row/col = +lane-16, K = 8..15 then 24..31
    const int mlane  = lane & 15;
    const int krlane = (lane >> 4) * 8;

    v8f acc[4][4];
    #pragma unroll
    for (int mt = 0; mt < 4; ++mt)
        #pragma unroll
        for (int nt = 0; nt < 4; ++nt)
            acc[mt][nt] = (v8f){0.f, 0.f, 0.f, 0.f, 0.f, 0.f, 0.f, 0.f};

    load_tiles(0);
    stage_to_lds(0);
    __syncthreads();

    for (int kc = 0; kc < KTILES; ++kc) {
        const int cur = kc & 1;
        if (kc + 1 < KTILES) load_tiles(kc + 1);   // issue next global loads first

        Frag ah[4], al[4];
        #pragma unroll
        for (int mt = 0; mt < 4; ++mt) {
            const int off = (wm * 64 + mt * 16 + mlane) * LDE + krlane;
            ah[mt].h[0] = *(const v8bf*)&Ah[cur][off];
            ah[mt].h[1] = *(const v8bf*)&Ah[cur][off + 16];
            al[mt].h[0] = *(const v8bf*)&Al[cur][off];
            al[mt].h[1] = *(const v8bf*)&Al[cur][off + 16];
        }
        #pragma unroll
        for (int nt = 0; nt < 4; ++nt) {           // one B hi/lo pair live at a time
            Frag bhf, blf;
            const int off = (wn * 64 + nt * 16 + mlane) * LDE + krlane;
            bhf.h[0] = *(const v8bf*)&Bh[cur][off];
            bhf.h[1] = *(const v8bf*)&Bh[cur][off + 16];
            blf.h[0] = *(const v8bf*)&Bl[cur][off];
            blf.h[1] = *(const v8bf*)&Bl[cur][off + 16];
            #pragma unroll
            for (int mt = 0; mt < 4; ++mt) {
                acc[mt][nt] = __builtin_amdgcn_wmma_f32_16x16x32_bf16(
                    false, ah[mt].v, false, bhf.v, (short)0, acc[mt][nt], false, false);
                acc[mt][nt] = __builtin_amdgcn_wmma_f32_16x16x32_bf16(
                    false, ah[mt].v, false, blf.v, (short)0, acc[mt][nt], false, false);
                acc[mt][nt] = __builtin_amdgcn_wmma_f32_16x16x32_bf16(
                    false, al[mt].v, false, bhf.v, (short)0, acc[mt][nt], false, false);
            }
        }

        if (kc + 1 < KTILES) stage_to_lds(cur ^ 1);  // disjoint buffer: no hazard w/ compute
        __syncthreads();                              // single barrier per K-chunk
    }

    // ---- epilogue: bias add + store (C/D layout: VGPR i, lane l -> M=i+8*(l>=16), N=l&15) ----
    #pragma unroll
    for (int mt = 0; mt < 4; ++mt) {
        const int rbase = m0 + wm * 64 + mt * 16 + (lane >> 4) * 8;
        #pragma unroll
        for (int nt = 0; nt < 4; ++nt) {
            const int n = n0 + wn * 64 + nt * 16 + mlane;
            const float bias = Bias[n];
            v8f a = acc[mt][nt];
            #pragma unroll
            for (int i = 0; i < 8; ++i)
                C[(size_t)(rbase + i) * N_DIM + n] = a[i] + bias;
        }
    }
}

extern "C" void kernel_launch(void* const* d_in, const int* in_sizes, int n_in,
                              void* d_out, int out_size, void* d_ws, size_t ws_size,
                              hipStream_t stream) {
    const float* x = (const float*)d_in[0];
    const float* W = (const float*)d_in[1];
    const float* b = (const float*)d_in[2];
    float* out = (float*)d_out;
    dim3 grid(N_DIM / BN, M_DIM / BM);  // 32 x 32 workgroups
    gemm_bf16x3_kernel<<<grid, 256, 0, stream>>>(x, W, b, out);
}
